// CEBlock_41678362640478
// MI455X (gfx1250) — compile-verified, matrix operand order checked by reading, not run on hardware
//
#include <hip/hip_runtime.h>
#include <hip/hip_bf16.h>
#include <math.h>

typedef __attribute__((ext_vector_type(16))) _Float16 v16h;
typedef __attribute__((ext_vector_type(8)))  _Float16 v8h;
typedef __attribute__((ext_vector_type(8)))  float    v8f;

// Problem constants
#define B_   32
#define N_   320
#define C_   768
#define H_   12
#define HD_  64
#define LT_  64
#define LS_  256
#define MLP_ 3072
#define BN_TOT (B_ * N_)   // 10240 rows

// GEMM tiling: 64x64 block tile, K-tile 64, 4 waves (128 thr), 32x32 per wave
#define BM 64
#define BN 64
#define BK 64
#define LDSS 72   // padded LDS row stride in halves (144B) -> bank-conflict-free

// ---------------------------------------------------------------------------
// Async direct global->LDS copy (CDNA5): tracked by ASYNCcnt, no VGPR bounce.
// ---------------------------------------------------------------------------
__device__ inline void async_ld16(_Float16* lds, const _Float16* gptr) {
    unsigned l = (unsigned)(uintptr_t)lds;                 // low 32b = LDS addr
    unsigned long long g = (unsigned long long)(uintptr_t)gptr;
    asm volatile("global_load_async_to_lds_b128 %0, %1, off"
                 :: "v"(l), "v"(g) : "memory");
}

// ---------------------------------------------------------------------------
// Fragment load: CDNA5 16-bit A/B layout. lane = row (m or n), hi = lane>>4.
// halves 0..7  -> K = hi*8 + 0..7 ; halves 8..15 -> K = 16 + hi*8 + 0..7
// => two contiguous 16B ds_load_b128 from a row-major (row x K) LDS tile.
// ---------------------------------------------------------------------------
__device__ inline v16h load_frag(const _Float16* __restrict__ rowp, int hi) {
    union { v16h v; v8h h[2]; } u;
    u.h[0] = *(const v8h*)(rowp + hi * 8);
    u.h[1] = *(const v8h*)(rowp + 16 + hi * 8);
    return u.v;
}

// ---------------------------------------------------------------------------
// Generic batched WMMA GEMM:  C[z] = epilogue( A[z] (MxK) * B[z] (KxN) )
// Bt is B transposed, row-major (N x K), f16. Grid: (N/64, M/64, nBatch).
// Batch offset: z -> (z/innerCnt)*outer + (z%innerCnt)*inner  per tensor.
// Double-buffered LDS, one-tile-ahead async prefetch.
// ---------------------------------------------------------------------------
__global__ __launch_bounds__(128)
void wmma_gemm_f16(const _Float16* __restrict__ A,
                   const _Float16* __restrict__ Bt,
                   float* __restrict__ C32,
                   _Float16* __restrict__ C16,
                   const float* __restrict__ bias,
                   const float* __restrict__ res,
                   int K, int lda, int ldb, int ldc,
                   long aOut, long aIn, long bOut, long bIn,
                   long cOut, long cIn, long rOut, long rIn,
                   int innerCnt, float alpha, int doGelu)
{
    __shared__ __align__(16) _Float16 sA[2][BM * LDSS];
    __shared__ __align__(16) _Float16 sB[2][BN * LDSS];

    const int z  = blockIdx.z;
    const int zo = z / innerCnt;
    const int zi = z - zo * innerCnt;
    A  += (long)zo * aOut + (long)zi * aIn;
    Bt += (long)zo * bOut + (long)zi * bIn;
    const long cOff = (long)zo * cOut + (long)zi * cIn;
    const long rOff = (long)zo * rOut + (long)zi * rIn;

    const int m0 = blockIdx.y * BM;
    const int n0 = blockIdx.x * BN;

    const int t    = threadIdx.x;
    const int lane = t & 31;
    const int wave = t >> 5;          // 0..3
    const int wm   = wave >> 1;       // 0..1 -> 32-row slab
    const int wn   = wave & 1;        // 0..1 -> 32-col slab
    const int hi   = lane >> 4;
    const int l15  = lane & 15;

    // Staging: 64 rows x 4 x 16B chunks per tile; thread t owns rows r0,r0+32
    const int r0 = t >> 2;            // 0..31
    const int cc = (t & 3) * 16;      // half offset within row

    const _Float16* gA = A  + (long)(m0 + r0) * lda + cc;
    const _Float16* gB = Bt + (long)(n0 + r0) * ldb + cc;

    auto issue = [&](int tile, int buf) {
        const long k0 = (long)tile * BK;
        _Float16* la = &sA[buf][r0 * LDSS + cc];
        _Float16* lb = &sB[buf][r0 * LDSS + cc];
        async_ld16(la,             gA + k0);
        async_ld16(la + 32 * LDSS, gA + k0 + 32L * lda);
        async_ld16(lb,             gB + k0);
        async_ld16(lb + 32 * LDSS, gB + k0 + 32L * ldb);
    };

    v8f acc00 = {}, acc01 = {}, acc10 = {}, acc11 = {};

    const int nTiles = K / BK;
    issue(0, 0);
    for (int i = 0; i < nTiles; ++i) {
        if (i + 1 < nTiles) {
            issue(i + 1, (i + 1) & 1);
            asm volatile("s_wait_asynccnt 0x4" ::: "memory"); // current tile landed
        } else {
            asm volatile("s_wait_asynccnt 0x0" ::: "memory");
        }
        __syncthreads();

        const _Float16* ab = sA[i & 1];
        const _Float16* bb = sB[i & 1];
        #pragma unroll
        for (int kk = 0; kk < 2; ++kk) {
            v16h a0 = load_frag(ab + (wm * 32      + l15) * LDSS + kk * 32, hi);
            v16h a1 = load_frag(ab + (wm * 32 + 16 + l15) * LDSS + kk * 32, hi);
            v16h b0 = load_frag(bb + (wn * 32      + l15) * LDSS + kk * 32, hi);
            v16h b1 = load_frag(bb + (wn * 32 + 16 + l15) * LDSS + kk * 32, hi);
            acc00 = __builtin_amdgcn_wmma_f32_16x16x32_f16(false, a0, false, b0, (short)0, acc00, false, false);
            acc01 = __builtin_amdgcn_wmma_f32_16x16x32_f16(false, a0, false, b1, (short)0, acc01, false, false);
            acc10 = __builtin_amdgcn_wmma_f32_16x16x32_f16(false, a1, false, b0, (short)0, acc10, false, false);
            acc11 = __builtin_amdgcn_wmma_f32_16x16x32_f16(false, a1, false, b1, (short)0, acc11, false, false);
        }
        __syncthreads();
    }

    // Epilogue. C/D layout: lane holds column n=l15, rows v + hi*8 (v=0..7)
    auto store_tile = [&](v8f acc, int mbase, int nbase) {
        const int n  = nbase + l15;
        const float bv = bias ? bias[n] : 0.0f;
        #pragma unroll
        for (int v = 0; v < 8; ++v) {
            const int m = mbase + v + hi * 8;
            float val = acc[v] * alpha + bv;
            if (res)    val += res[rOff + (long)m * ldc + n];
            if (doGelu) val = 0.5f * val * (1.0f + erff(val * 0.70710678118654752f));
            if (C32) C32[cOff + (long)m * ldc + n] = val;
            if (C16) C16[cOff + (long)m * ldc + n] = (_Float16)val;
        }
    };
    const int mb = m0 + wm * 32;
    const int nb = n0 + wn * 32;
    store_tile(acc00, mb,      nb);
    store_tile(acc01, mb,      nb + 16);
    store_tile(acc10, mb + 16, nb);
    store_tile(acc11, mb + 16, nb + 16);
}

// ---------------------------------------------------------------------------
// Weight convert + transpose: W (rows x cols) f32 -> WT (cols x rows) f16
// ---------------------------------------------------------------------------
__global__ __launch_bounds__(256)
void convert_transpose(const float* __restrict__ W, _Float16* __restrict__ WT,
                       int rows, int cols)
{
    long i = (long)blockIdx.x * 256 + threadIdx.x;
    if (i >= (long)rows * cols) return;
    long r = i / cols;
    int  c = (int)(i - r * cols);
    WT[(long)c * rows + r] = (_Float16)W[i];
}

// ---------------------------------------------------------------------------
// LayerNorm over C=768, output f16. One 256-thread block per row.
// ---------------------------------------------------------------------------
__global__ __launch_bounds__(256)
void ln_f16(const float* __restrict__ x, const float* __restrict__ g,
            const float* __restrict__ bta, _Float16* __restrict__ out)
{
    const int row = blockIdx.x;
    const float* xr = x + (long)row * C_;
    float v[3], s = 0.f, ss = 0.f;
    #pragma unroll
    for (int j = 0; j < 3; ++j) {
        v[j] = xr[threadIdx.x + j * 256];
        s += v[j]; ss += v[j] * v[j];
    }
    __shared__ float shs[8], shq[8];
    #pragma unroll
    for (int o = 16; o > 0; o >>= 1) { s += __shfl_xor(s, o); ss += __shfl_xor(ss, o); }
    if ((threadIdx.x & 31) == 0) { shs[threadIdx.x >> 5] = s; shq[threadIdx.x >> 5] = ss; }
    __syncthreads();
    float S = 0.f, SS = 0.f;
    #pragma unroll
    for (int i = 0; i < 8; ++i) { S += shs[i]; SS += shq[i]; }
    const float mean = S * (1.0f / C_);
    const float rstd = rsqrtf(SS * (1.0f / C_) - mean * mean + 1e-5f);
    #pragma unroll
    for (int j = 0; j < 3; ++j) {
        int c = threadIdx.x + j * 256;
        out[(long)row * C_ + c] = (_Float16)((v[j] - mean) * rstd * g[c] + bta[c]);
    }
}

// Final: x = x + LN(x)*g + b, in place, f32
__global__ __launch_bounds__(256)
void final_ln_add(float* __restrict__ x, const float* __restrict__ g,
                  const float* __restrict__ bta)
{
    const int row = blockIdx.x;
    float* xr = x + (long)row * C_;
    float v[3], s = 0.f, ss = 0.f;
    #pragma unroll
    for (int j = 0; j < 3; ++j) {
        v[j] = xr[threadIdx.x + j * 256];
        s += v[j]; ss += v[j] * v[j];
    }
    __shared__ float shs[8], shq[8];
    #pragma unroll
    for (int o = 16; o > 0; o >>= 1) { s += __shfl_xor(s, o); ss += __shfl_xor(ss, o); }
    if ((threadIdx.x & 31) == 0) { shs[threadIdx.x >> 5] = s; shq[threadIdx.x >> 5] = ss; }
    __syncthreads();
    float S = 0.f, SS = 0.f;
    #pragma unroll
    for (int i = 0; i < 8; ++i) { S += shs[i]; SS += shq[i]; }
    const float mean = S * (1.0f / C_);
    const float rstd = rsqrtf(SS * (1.0f / C_) - mean * mean + 1e-5f);
    #pragma unroll
    for (int j = 0; j < 3; ++j) {
        int c = threadIdx.x + j * 256;
        xr[c] = v[j] + (v[j] - mean) * rstd * g[c] + bta[c];
    }
}

// ---------------------------------------------------------------------------
// Split qkv f32 (B,N,3,H,HD) -> q16/k16 (B,H,N,HD) and vT16 (B,H,HD,N)
// ---------------------------------------------------------------------------
__global__ __launch_bounds__(256)
void split_qkv(const float* __restrict__ qkv, _Float16* __restrict__ q16,
               _Float16* __restrict__ k16, _Float16* __restrict__ vT16)
{
    long i = (long)blockIdx.x * 256 + threadIdx.x;
    if (i >= (long)BN_TOT * C_) return;
    int  c  = (int)(i % C_);
    long bn = i / C_;
    int  n  = (int)(bn % N_);
    int  b  = (int)(bn / N_);
    int  h  = c >> 6;
    int  d  = c & 63;
    const float* base = qkv + bn * (3L * C_);
    long qi = (((long)b * H_ + h) * N_ + n) * HD_ + d;
    q16[qi] = (_Float16)base[c];
    k16[qi] = (_Float16)base[C_ + c];
    vT16[(((long)b * H_ + h) * HD_ + d) * N_ + n] = (_Float16)base[2 * C_ + c];
}

// ---------------------------------------------------------------------------
// Row softmax over 320 logits; normalizes f32 in place + writes f16 copy.
// ---------------------------------------------------------------------------
__global__ __launch_bounds__(128)
void softmax320(float* __restrict__ logits, _Float16* __restrict__ attn16)
{
    const long row = blockIdx.x;
    float* lr = logits + row * N_;
    const int tid = threadIdx.x;
    __shared__ float sh[4];
    float mx = -3.4e38f;
    for (int c = tid; c < N_; c += 128) mx = fmaxf(mx, lr[c]);
    #pragma unroll
    for (int o = 16; o > 0; o >>= 1) mx = fmaxf(mx, __shfl_xor(mx, o));
    if ((tid & 31) == 0) sh[tid >> 5] = mx;
    __syncthreads();
    mx = fmaxf(fmaxf(sh[0], sh[1]), fmaxf(sh[2], sh[3]));
    __syncthreads();
    float s = 0.f;
    for (int c = tid; c < N_; c += 128) { float e = __expf(lr[c] - mx); lr[c] = e; s += e; }
    #pragma unroll
    for (int o = 16; o > 0; o >>= 1) s += __shfl_xor(s, o);
    if ((tid & 31) == 0) sh[tid >> 5] = s;
    __syncthreads();
    const float inv = 1.0f / (sh[0] + sh[1] + sh[2] + sh[3]);
    for (int c = tid; c < N_; c += 128) {
        float a = lr[c] * inv;
        lr[c] = a;
        attn16[row * N_ + c] = (_Float16)a;
    }
}

// ---------------------------------------------------------------------------
// w_ts = softmax_s( mean_h attn[b,h,t, LT+s] ). One block per (b,t), s=tid.
// ---------------------------------------------------------------------------
__global__ __launch_bounds__(256)
void wts_softmax(const float* __restrict__ attn, float* __restrict__ wts)
{
    const int bt = blockIdx.x;
    const int b  = bt >> 6;
    const int t  = bt & 63;
    const int s  = threadIdx.x;
    float v = 0.f;
    #pragma unroll
    for (int h = 0; h < H_; ++h)
        v += attn[(((long)b * H_ + h) * N_ + t) * N_ + LT_ + s];
    v *= (1.0f / H_);
    __shared__ float sh[8];
    float mx = v;
    #pragma unroll
    for (int o = 16; o > 0; o >>= 1) mx = fmaxf(mx, __shfl_xor(mx, o));
    if ((s & 31) == 0) sh[s >> 5] = mx;
    __syncthreads();
    mx = sh[0];
    #pragma unroll
    for (int i = 1; i < 8; ++i) mx = fmaxf(mx, sh[i]);
    __syncthreads();
    const float e = __expf(v - mx);
    float sum = e;
    #pragma unroll
    for (int o = 16; o > 0; o >>= 1) sum += __shfl_xor(sum, o);
    if ((s & 31) == 0) sh[s >> 5] = sum;
    __syncthreads();
    float tot = 0.f;
    #pragma unroll
    for (int i = 0; i < 8; ++i) tot += sh[i];
    wts[(long)bt * LS_ + s] = e / tot;
}

// ---------------------------------------------------------------------------
// Sparse adjacency matmul: out = adj @ support (block structure).
// One block per (b,n); 256 threads cover 768 columns (3 each).
// ---------------------------------------------------------------------------
__global__ __launch_bounds__(256)
void adj_mm(const float* __restrict__ wts, const float* __restrict__ sup,
            float* __restrict__ out)
{
    const int bn = blockIdx.x;
    const int b  = bn / N_;
    const int n  = bn - b * N_;
    const int c0 = threadIdx.x, c1 = c0 + 256, c2 = c0 + 512;
    float a0 = 0.f, a1 = 0.f, a2 = 0.f;
    if (n < LT_) {
        const float* wr = wts + ((long)b * LT_ + n) * LS_;
        const float* sb = sup + ((long)b * N_ + LT_) * C_;
        for (int s = 0; s < LS_; ++s) {
            float w = wr[s];
            const float* sp = sb + (long)s * C_;
            a0 += w * sp[c0]; a1 += w * sp[c1]; a2 += w * sp[c2];
        }
    } else {
        const int s = n - LT_;
        const float* sb = sup + (long)b * N_ * C_;
        for (int t = 0; t < LT_; ++t) {
            float w = wts[((long)b * LT_ + t) * LS_ + s];
            const float* sp = sb + (long)t * C_;
            a0 += w * sp[c0]; a1 += w * sp[c1]; a2 += w * sp[c2];
        }
    }
    float* o = out + (long)bn * C_;
    o[c0] = a0; o[c1] = a1; o[c2] = a2;
}

// ---------------------------------------------------------------------------
extern "C" void kernel_launch(void* const* d_in, const int* in_sizes, int n_in,
                              void* d_out, int out_size, void* d_ws, size_t ws_size,
                              hipStream_t stream)
{
    (void)in_sizes; (void)n_in; (void)out_size; (void)ws_size;

    const float* x_in  = (const float*)d_in[0];
    const float* g1    = (const float*)d_in[3];
    const float* b1    = (const float*)d_in[4];
    const float* Wqkv  = (const float*)d_in[5];
    const float* Wproj = (const float*)d_in[6];
    const float* bproj = (const float*)d_in[7];
    const float* g2    = (const float*)d_in[8];
    const float* b2    = (const float*)d_in[9];
    const float* W1    = (const float*)d_in[10];
    const float* bm1   = (const float*)d_in[11];
    const float* W2    = (const float*)d_in[12];
    const float* bm2   = (const float*)d_in[13];
    const float* Wg    = (const float*)d_in[14];
    const float* bg    = (const float*)d_in[15];
    const float* g3    = (const float*)d_in[16];
    const float* b3    = (const float*)d_in[17];
    float* out = (float*)d_out;

    // workspace carve-out (256B aligned)
    char* ws = (char*)d_ws;
    size_t off = 0;
    auto alloc = [&](size_t bytes) -> void* {
        void* p = ws + off;
        off = (off + bytes + 255) & ~(size_t)255;
        return p;
    };
    _Float16* wqkvT  = (_Float16*)alloc((size_t)3 * C_ * C_ * 2);       // 2304x768
    _Float16* wprojT = (_Float16*)alloc((size_t)C_ * C_ * 2);           // 768x768
    _Float16* w1T    = (_Float16*)alloc((size_t)MLP_ * C_ * 2);         // 3072x768
    _Float16* w2T    = (_Float16*)alloc((size_t)C_ * MLP_ * 2);         // 768x3072
    _Float16* wgT    = (_Float16*)alloc((size_t)C_ * C_ * 2);
    _Float16* xn16   = (_Float16*)alloc((size_t)BN_TOT * C_ * 2);
    float*    qkv32  = (float*)   alloc((size_t)BN_TOT * 3 * C_ * 4);
    _Float16* q16    = (_Float16*)alloc((size_t)B_ * H_ * N_ * HD_ * 2);
    _Float16* k16    = (_Float16*)alloc((size_t)B_ * H_ * N_ * HD_ * 2);
    _Float16* vT16   = (_Float16*)alloc((size_t)B_ * H_ * HD_ * N_ * 2);
    float*    logits = (float*)   alloc((size_t)B_ * H_ * N_ * N_ * 4); // -> attn f32 in place
    _Float16* attn16 = (_Float16*)alloc((size_t)B_ * H_ * N_ * N_ * 2);
    _Float16* ctx16  = (_Float16*)alloc((size_t)BN_TOT * C_ * 2);
    float*    x1     = (float*)   alloc((size_t)BN_TOT * C_ * 4);
    _Float16* h16    = (_Float16*)alloc((size_t)BN_TOT * C_ * 2);
    _Float16* mlp16  = (_Float16*)alloc((size_t)BN_TOT * MLP_ * 2);
    _Float16* x216   = (_Float16*)alloc((size_t)BN_TOT * C_ * 2);
    float*    sup32  = (float*)   alloc((size_t)BN_TOT * C_ * 4);
    float*    wts32  = (float*)   alloc((size_t)B_ * LT_ * LS_ * 4);

    const float ascale = 0.125f; // HD^-0.5 = 1/8

    // 1) weights -> transposed f16
    auto cgrid = [](long e) { return (unsigned)((e + 255) / 256); };
    convert_transpose<<<cgrid((long)C_ * 3 * C_), 256, 0, stream>>>(Wqkv,  wqkvT,  C_,  3 * C_);
    convert_transpose<<<cgrid((long)C_ * C_),     256, 0, stream>>>(Wproj, wprojT, C_,  C_);
    convert_transpose<<<cgrid((long)C_ * MLP_),   256, 0, stream>>>(W1,    w1T,    C_,  MLP_);
    convert_transpose<<<cgrid((long)MLP_ * C_),   256, 0, stream>>>(W2,    w2T,    MLP_, C_);
    convert_transpose<<<cgrid((long)C_ * C_),     256, 0, stream>>>(Wg,    wgT,    C_,  C_);

    // 2) LN1 -> f16
    ln_f16<<<BN_TOT, 256, 0, stream>>>(x_in, g1, b1, xn16);

    // 3) QKV GEMM: (10240x768) x (768x2304) -> f32
    wmma_gemm_f16<<<dim3(3 * C_ / BN, BN_TOT / BM, 1), 128, 0, stream>>>(
        xn16, wqkvT, qkv32, nullptr, nullptr, nullptr,
        C_, C_, C_, 3 * C_, 0, 0, 0, 0, 0, 0, 0, 0, 1, 1.0f, 0);

    // 4) split into per-head f16 operands
    split_qkv<<<cgrid((long)BN_TOT * C_), 256, 0, stream>>>(qkv32, q16, k16, vT16);

    // 5) logits = scale * Q Kᵀ, batched over B*H (K=64)
    wmma_gemm_f16<<<dim3(N_ / BN, N_ / BM, B_ * H_), 128, 0, stream>>>(
        q16, k16, logits, nullptr, nullptr, nullptr,
        HD_, HD_, HD_, N_,
        (long)N_ * HD_, 0, (long)N_ * HD_, 0, (long)N_ * N_, 0, 0, 0,
        1, ascale, 0);

    // 6) softmax rows (attn f32 in place + f16 copy)
    softmax320<<<B_ * H_ * N_, 128, 0, stream>>>(logits, attn16);

    // 7) w_ts head-mean + softmax
    wts_softmax<<<B_ * LT_, 256, 0, stream>>>(logits, wts32);

    // 8) ctx = attn @ V, batched; scatter f16 into (B,N,C) per head
    wmma_gemm_f16<<<dim3(HD_ / BN, N_ / BM, B_ * H_), 128, 0, stream>>>(
        attn16, vT16, nullptr, ctx16, nullptr, nullptr,
        N_, N_, N_, C_,
        (long)H_ * N_ * N_, (long)N_ * N_,
        (long)H_ * HD_ * N_, (long)HD_ * N_,
        (long)N_ * C_, (long)HD_, 0, 0,
        H_, 1.0f, 0);

    // 9) x1 = x + ctx @ Wproj + bproj
    wmma_gemm_f16<<<dim3(C_ / BN, BN_TOT / BM, 1), 128, 0, stream>>>(
        ctx16, wprojT, x1, nullptr, bproj, x_in,
        C_, C_, C_, C_, 0, 0, 0, 0, 0, 0, 0, 0, 1, 1.0f, 0);

    // 10) LN2 -> f16
    ln_f16<<<BN_TOT, 256, 0, stream>>>(x1, g2, b2, h16);

    // 11) MLP1: GELU(h @ W1 + bm1) -> f16
    wmma_gemm_f16<<<dim3(MLP_ / BN, BN_TOT / BM, 1), 128, 0, stream>>>(
        h16, w1T, nullptr, mlp16, bm1, nullptr,
        C_, C_, C_, MLP_, 0, 0, 0, 0, 0, 0, 0, 0, 1, 1.0f, 1);

    // 12) x2 = x1 + mlp @ W2 + bm2 -> f16
    wmma_gemm_f16<<<dim3(C_ / BN, BN_TOT / BM, 1), 128, 0, stream>>>(
        mlp16, w2T, nullptr, x216, bm2, x1,
        MLP_, MLP_, MLP_, C_, 0, 0, 0, 0, 0, 0, 0, 0, 1, 1.0f, 0);

    // 13) support = x2 @ Wg + bg -> f32
    wmma_gemm_f16<<<dim3(C_ / BN, BN_TOT / BM, 1), 128, 0, stream>>>(
        x216, wgT, sup32, nullptr, bg, nullptr,
        C_, C_, C_, C_, 0, 0, 0, 0, 0, 0, 0, 0, 1, 1.0f, 0);

    // 14) x3 = adj @ support  (block-sparse)
    adj_mm<<<BN_TOT, 256, 0, stream>>>(wts32, sup32, out);

    // 15) out = x3 + LN(x3)*g3 + b3  (in place)
    final_ln_add<<<BN_TOT, 256, 0, stream>>>(out, g3, b3);
}
